// Conv2d_lsq_int_83425444758206
// MI455X (gfx1250) — 8-bit matrix operand order repaired and checked by reading, compiled, not run on hardware
//
#include <hip/hip_runtime.h>
#include <hip/hip_bf16.h>


typedef int  v8i  __attribute__((ext_vector_type(8)));

static constexpr int B_ = 16, C_ = 320, H_ = 64, W_ = 64;
static constexpr size_t XQ_BYTES = (size_t)B_ * H_ * W_ * C_;
static constexpr size_t WQ_BYTES = (size_t)9 * C_ * C_;
static constexpr size_t WQ_OFF = XQ_BYTES;
static constexpr size_t BQ_OFF = XQ_BYTES + WQ_BYTES;

static constexpr int PIX_STRIDE = 80;

__global__ void quant_x_kernel(const float* __restrict__ x,
                               const float* __restrict__ step_x,
                               signed char* __restrict__ xq, int n) {
  int idx = blockIdx.x * blockDim.x + threadIdx.x;
  if (idx >= n) return;
  float inv = 1.0f / step_x[0];
  int w  = idx & 63;
  int h  = (idx >> 6) & 63;
  int ci = (idx >> 12) % C_;
  int b  = idx / (C_ * H_ * W_);
  float q = rintf(x[idx] * inv);
  q = fmaxf(-127.f, fminf(127.f, q));
  xq[(((size_t)(b * 64 + h) * 64 + w) * C_) + ci] = (signed char)(int)q;
}

__global__ void quant_w_kernel(const float* __restrict__ wt,
                               const float* __restrict__ step_w,
                               signed char* __restrict__ wq, int n) {
  int idx = blockIdx.x * blockDim.x + threadIdx.x;
  if (idx >= n) return;
  float inv = 1.0f / step_w[0];
  int k   = idx % 9;
  int ci  = (idx / 9) % C_;
  int co  = idx / (9 * C_);
  float q = rintf(wt[idx] * inv);
  q = fmaxf(-127.f, fminf(127.f, q));
  wq[((size_t)k * C_ + co) * C_ + ci] = (signed char)(int)q;
}

__global__ void quant_bias_kernel(const float* __restrict__ bias,
                                  const float* __restrict__ step_b,
                                  const float* __restrict__ step_x,
                                  const float* __restrict__ step_w,
                                  const float* __restrict__ shift_scale,
                                  float* __restrict__ bq) {
  int c = blockIdx.x * blockDim.x + threadIdx.x;
  if (c >= C_) return;
  float sb = step_b[0];
  float q = rintf(bias[c] / sb);
  q = fmaxf(-127.f, fminf(127.f, q));
  float bdeq = q * sb;
  float scale = shift_scale[0] / (step_x[0] * step_w[0]);
  float bi = rintf(bdeq * scale);
  bi = fmaxf(-127.f, fminf(127.f, bi));
  bq[c] = bi;
}

__global__ __launch_bounds__(256) void conv_wmma_kernel(
    const signed char* __restrict__ xq,
    const signed char* __restrict__ wq,
    const float* __restrict__ bq,
    const float* __restrict__ shift_scale,
    float* __restrict__ out) {
  __shared__ signed char xs[4 * 66 * PIX_STRIDE];

  const int tid  = threadIdx.x;
  const int lane = tid & 31;
  const int wave = tid >> 5;

  const int cb   = blockIdx.x & 3;
  const int hblk = (blockIdx.x >> 2) & 31;
  const int b    = blockIdx.x >> 7;

  const int h0      = hblk * 2;
  const int co_blk  = cb * 80;
  const int hsub    = wave >> 2;
  const int wtile   = wave & 3;

  const int ln   = lane & 15;
  const int half = lane >> 4;

  v8i acc[5];
#pragma unroll
  for (int t = 0; t < 5; ++t) acc[t] = v8i{};

  for (int c = 0; c < 5; ++c) {
    const int cioff = c * 64;

    __syncthreads();
    for (int s = tid; s < 4 * 66; s += 256) {
      const int row  = s / 66;
      const int pix  = s % 66;
      const int h_in = h0 - 1 + row;
      const int w_in = pix - 1;
      const bool valid = ((unsigned)h_in < 64u) && ((unsigned)w_in < 64u);
      const int h_c = min(63, max(0, h_in));
      const int w_c = min(63, max(0, w_in));
      const int4* src = (const int4*)(xq +
          ((size_t)((b * 64 + h_c) * 64 + w_c) * C_) + cioff);
      const int4 z = {0, 0, 0, 0};
      int4 v0 = valid ? src[0] : z;
      int4 v1 = valid ? src[1] : z;
      int4 v2 = valid ? src[2] : z;
      int4 v3 = valid ? src[3] : z;
      int4* dst = (int4*)(&xs[(size_t)s * PIX_STRIDE]);
      dst[0] = v0; dst[1] = v1; dst[2] = v2; dst[3] = v3;
    }
    __syncthreads();

#pragma unroll
    for (int kh = 0; kh < 3; ++kh) {
#pragma unroll
      for (int kw = 0; kw < 3; ++kw) {
        const signed char* ab =
            &xs[(size_t)((hsub + kh) * 66 + (wtile * 16 + ln + kw)) * PIX_STRIDE];
        v8i a;
#pragma unroll
        for (int v = 0; v < 8; ++v) {
          const int koffA = 32 * (v >> 2) + 16 * ((v >> 1) & 1) + 8 * half + 4 * (v & 1);
          a[v] = *(const int*)(ab + koffA);
        }
        const size_t tap = (size_t)(kh * 3 + kw);
#pragma unroll
        for (int t = 0; t < 5; ++t) {
          const int co = co_blk + t * 16 + ln;
          const signed char* bb = wq + (tap * C_ + co) * C_ + cioff;
          v8i bm;
#pragma unroll
          for (int v = 0; v < 8; ++v) {
            const int koffB = 32 * (v >> 2) + 16 * ((v >> 1) & 1) + 8 * half + 4 * (v & 1);
            bm[v] = *(const int*)(bb + koffB);
          }
          acc[t] = __builtin_amdgcn_wmma_i32_16x16x64_iu8(
               true, a,  true, bm, acc[t],
               false,  false);
        }
      }
    }
  }

  const float shift = shift_scale[0];
  const int h = h0 + hsub;
#pragma unroll
  for (int t = 0; t < 5; ++t) {
    const int co = co_blk + t * 16 + ln;
    const float bi = bq[co];
#pragma unroll
    for (int r = 0; r < 8; ++r) {
      const int m = r + 8 * half;
      float y = rintf((float)acc[t][r] * shift);
      y = fmaxf(-127.f, fminf(127.f, y));
      y = y + bi;
      y = fmaxf(-127.f, fminf(127.f, y));
      const size_t o = ((size_t)(b * C_ + co) * 64 + h) * 64 + (wtile * 16 + m);
      out[o] = y;
    }
  }
}

extern "C" void kernel_launch(void* const* d_in, const int* in_sizes, int n_in,
                              void* d_out, int out_size, void* d_ws, size_t ws_size,
                              hipStream_t stream) {
  const float* x           = (const float*)d_in[0];
  const float* weight      = (const float*)d_in[1];
  const float* bias        = (const float*)d_in[2];
  const float* step_x      = (const float*)d_in[3];
  const float* step_w      = (const float*)d_in[4];
  const float* step_b      = (const float*)d_in[5];
  const float* shift_scale = (const float*)d_in[6];

  signed char* xq = (signed char*)d_ws;
  signed char* wq = (signed char*)d_ws + WQ_OFF;
  float*       bq = (float*)((char*)d_ws + BQ_OFF);
  float*       out = (float*)d_out;

  const int nx = B_ * C_ * H_ * W_;
  const int nw = C_ * C_ * 9;

  quant_x_kernel<<<(nx + 255) / 256, 256, 0, stream>>>(x, step_x, xq, nx);
  quant_w_kernel<<<(nw + 255) / 256, 256, 0, stream>>>(weight, step_w, wq, nw);
  quant_bias_kernel<<<2, 256, 0, stream>>>(bias, step_b, step_x, step_w,
                                           shift_scale, bq);
  conv_wmma_kernel<<<B_ * 32 * 4, 256, 0, stream>>>(xq, wq, bq, shift_scale, out);
}
